// GAEencoder_4071628996674
// MI455X (gfx1250) — matrix operand order repaired and checked by reading, not compiled, not run on hardware
//
#include <hip/hip_runtime.h>
#include <hip/hip_bf16.h>

typedef __attribute__((ext_vector_type(16))) __bf16         v16bf;
typedef __attribute__((ext_vector_type(16))) unsigned short v16u;
typedef __attribute__((ext_vector_type(8)))  unsigned short us8;
typedef __attribute__((ext_vector_type(8)))  float          v8f;

// ---------- helpers: fp32 -> bf16 (RNE) and back ----------
__device__ __forceinline__ unsigned short f2bf(float f) {
    unsigned u = __builtin_bit_cast(unsigned, f);
    u += 0x7FFFu + ((u >> 16) & 1u);
    return (unsigned short)(u >> 16);
}
__device__ __forceinline__ float bf2f(unsigned short b) {
    return __builtin_bit_cast(float, (unsigned)b << 16);
}

// ---------- small utility kernels ----------
__global__ void k_init(float* deg, float* stats, int M) {
    int i = blockIdx.x * blockDim.x + threadIdx.x;
    if (i < M)   deg[i]   = 1.0f;   // self-loop contributes 1 to degree
    if (i < 384) stats[i] = 0.0f;   // zero BN accumulators
}

__global__ void k_degree(const int* __restrict__ dst, float* __restrict__ deg, int E) {
    int i = blockIdx.x * blockDim.x + threadIdx.x;
    if (i < E) atomicAdd(&deg[dst[i]], 1.0f);
}

__global__ void k_dinv(float* deg, int M) {
    int i = blockIdx.x * blockDim.x + threadIdx.x;
    if (i < M) deg[i] = rsqrtf(deg[i]);   // deg >= 1 always
}

// A[i][:] = H[i][:] * dinv[i]^2  (self-loop term, no atomics needed)
template<int C>
__global__ void k_selfinit(const float* __restrict__ H, const float* __restrict__ dinv,
                           float* __restrict__ A, int M) {
    int i = blockIdx.x * blockDim.x + threadIdx.x;
    int total = M * (C / 4);
    if (i >= total) return;
    int r = i / (C / 4);
    int c = (i % (C / 4)) * 4;
    float w = dinv[r]; w *= w;
    const float4 h = *(const float4*)(H + (size_t)r * C + c);
    float4 o; o.x = h.x * w; o.y = h.y * w; o.z = h.z * w; o.w = h.w * w;
    *(float4*)(A + (size_t)r * C + c) = o;
}

// per-edge scatter: A[dst] += H[src] * dinv[src]*dinv[dst]
template<int C>
__global__ void k_scatter(const int* __restrict__ src, const int* __restrict__ dst,
                          const float* __restrict__ dinv, const float* __restrict__ H,
                          float* __restrict__ A, int E) {
    const int TPE = C / 4;                       // threads per edge (float4 each)
    long long i = (long long)blockIdx.x * blockDim.x + threadIdx.x;
    long long total = (long long)E * TPE;
    if (i >= total) return;
    int e = (int)(i / TPE);
    int c = (int)(i % TPE) * 4;
    int s = src[e], d = dst[e];
    float nrm = dinv[s] * dinv[d];
    const float4 h = *(const float4*)(H + (size_t)s * C + c);
    float* a = A + (size_t)d * C + c;
    atomicAdd(a + 0, h.x * nrm);
    atomicAdd(a + 1, h.y * nrm);
    atomicAdd(a + 2, h.z * nrm);
    atomicAdd(a + 3, h.w * nrm);
}

// per-channel sum / sum-of-squares over rows
template<int C>
__global__ void k_stats(const float* __restrict__ A, float* __restrict__ gsum,
                        float* __restrict__ gsq, int M) {
    const int TPC = 256 / C;
    int tid = threadIdx.x;
    int c  = tid & (C - 1);
    int r  = blockIdx.x * TPC + tid / C;
    int rs = gridDim.x * TPC;
    float s = 0.f, q = 0.f;
    for (; r < M; r += rs) {
        float v = A[(size_t)r * C + c];
        s += v; q += v * v;
    }
    atomicAdd(&gsum[c], s);
    atomicAdd(&gsq[c],  q);
}

// scale/shift for fused batchnorm: y = x*scale + shift
__global__ void k_bnfin(const float* __restrict__ gsum, const float* __restrict__ gsq,
                        const float* __restrict__ gamma, const float* __restrict__ beta,
                        float* __restrict__ scale, float* __restrict__ shift,
                        int C, float invM) {
    int c = threadIdx.x;
    if (c >= C) return;
    float mean = gsum[c] * invM;
    float var  = gsq[c] * invM - mean * mean;
    float sc   = gamma[c] * rsqrtf(var + 1e-5f);
    scale[c] = sc;
    shift[c] = beta[c] - mean * sc;
}

__global__ void k_out(const float* __restrict__ A2, const float* __restrict__ scale,
                      const float* __restrict__ shift, float* __restrict__ out, int M) {
    int i = blockIdx.x * blockDim.x + threadIdx.x;
    if (i >= M * 64) return;
    int c = i & 63;
    out[i] = fmaf(A2[i], scale[c], shift[c]);
}

// ---------- pre-conversion kernels (fast path) ----------
// fp32 [rows x 128] -> packed bf16 hi/lo (optionally fusing BN+ReLU, channel = i&127)
template<bool BN_RELU>
__global__ void k_cvtX(const float* __restrict__ X,
                       const float* __restrict__ scale, const float* __restrict__ shift,
                       unsigned short* __restrict__ hi, unsigned short* __restrict__ lo,
                       int total) {
    int i = blockIdx.x * blockDim.x + threadIdx.x;
    if (i >= total) return;
    float f = X[i];
    if (BN_RELU) {
        int c = i & 127;
        f = fmaxf(fmaf(f, scale[c], shift[c]), 0.0f);
    }
    unsigned short h = f2bf(f);
    hi[i] = h;
    lo[i] = f2bf(f - bf2f(h));
}

// W [128 x NCOLS] fp32 -> per-lane swizzled bf16 hi/lo fragments:
// frag[((ntile*4 + kstep)*32 + lane)*16 + j] = W[kstep*32 + (lane>>4)*8 + (j&7) + (j>>3)*16][ntile*16 + (lane&15)]
__global__ void k_cvtW(const float* __restrict__ W,
                       unsigned short* __restrict__ fh, unsigned short* __restrict__ fl,
                       int NCOLS) {
    int i = blockIdx.x * blockDim.x + threadIdx.x;
    int total = (NCOLS / 16) * 2048;
    if (i >= total) return;
    int j    = i & 15;
    int lane = (i >> 4) & 31;
    int s    = (i >> 9) & 3;
    int t    = i >> 11;
    int k = s * 32 + (lane >> 4) * 8 + (j & 7) + (j >> 3) * 16;
    int n = t * 16 + (lane & 15);
    float w = W[(size_t)k * NCOLS + n];
    unsigned short h = f2bf(w);
    fh[i] = h;
    fl[i] = f2bf(w - bf2f(h));
}

// ---------- WMMA GEMM, fast path: pure loads + WMMA ----------
// H[M x NCOLS] = X * W with X given as packed bf16 hi/lo [M x 128] and W as
// pre-swizzled fragments. bf16x3: Xhi*Whi + Xhi*Wlo + Xlo*Whi.
template<int NCOLS>
__global__ void k_gemm_fast(const unsigned short* __restrict__ Xhi,
                            const unsigned short* __restrict__ Xlo,
                            const unsigned short* __restrict__ Wfh,
                            const unsigned short* __restrict__ Wfl,
                            float* __restrict__ H, int M) {
    const int lane = threadIdx.x & 31;
    const int wv   = threadIdx.x >> 5;         // N-tile index
    const int half = lane >> 4;
    const int l16  = lane & 15;
    int row = blockIdx.x * 16 + l16;
    if (row >= M) row = M - 1;                 // clamp keeps EXEC uniform
    const unsigned short* xh = Xhi + (size_t)row * 128;
    const unsigned short* xl = Xlo + (size_t)row * 128;
    const unsigned short* bh = Wfh + ((size_t)(wv * 4) * 32 + lane) * 16;
    const unsigned short* bl = Wfl + ((size_t)(wv * 4) * 32 + lane) * 16;

    v8f acc = {};
    #pragma unroll
    for (int s = 0; s < 4; ++s) {
        const int ka = s * 32 + half * 8;
        us8 a0 = *(const us8*)(xh + ka);
        us8 a1 = *(const us8*)(xh + ka + 16);
        us8 a2 = *(const us8*)(xl + ka);
        us8 a3 = *(const us8*)(xl + ka + 16);
        us8 b0 = *(const us8*)(bh + s * 512);
        us8 b1 = *(const us8*)(bh + s * 512 + 8);
        us8 b2 = *(const us8*)(bl + s * 512);
        us8 b3 = *(const us8*)(bl + s * 512 + 8);
        v16u ahu = __builtin_shufflevector(a0, a1, 0,1,2,3,4,5,6,7,8,9,10,11,12,13,14,15);
        v16u alu = __builtin_shufflevector(a2, a3, 0,1,2,3,4,5,6,7,8,9,10,11,12,13,14,15);
        v16u bhu = __builtin_shufflevector(b0, b1, 0,1,2,3,4,5,6,7,8,9,10,11,12,13,14,15);
        v16u blu = __builtin_shufflevector(b2, b3, 0,1,2,3,4,5,6,7,8,9,10,11,12,13,14,15);
        v16bf ah = __builtin_bit_cast(v16bf, ahu);
        v16bf al = __builtin_bit_cast(v16bf, alu);
        v16bf bhv = __builtin_bit_cast(v16bf, bhu);
        v16bf blv = __builtin_bit_cast(v16bf, blu);
        acc = __builtin_amdgcn_wmma_f32_16x16x32_bf16(false, ah, false, bhv, (short)0, acc, false, false);
        acc = __builtin_amdgcn_wmma_f32_16x16x32_bf16(false, ah, false, blv, (short)0, acc, false, false);
        acc = __builtin_amdgcn_wmma_f32_16x16x32_bf16(false, al, false, bhv, (short)0, acc, false, false);
    }
    const int orow = blockIdx.x * 16 + half * 8;
    #pragma unroll
    for (int i = 0; i < 8; ++i) {
        int r = orow + i;
        if (r < M) H[(size_t)r * NCOLS + wv * 16 + l16] = acc[i];
    }
}

// ---------- WMMA GEMM, compact fallback: inline bf16 split ----------
template<int NCOLS, bool BN_RELU>
__global__ void k_gemm(const float* __restrict__ X, const float* __restrict__ W,
                       float* __restrict__ H,
                       const float* __restrict__ scale, const float* __restrict__ shift,
                       int M) {
    const int lane = threadIdx.x & 31;
    const int wv   = threadIdx.x >> 5;
    const int half = lane >> 4;
    const int l16  = lane & 15;
    int row  = blockIdx.x * 16 + l16;
    if (row >= M) row = M - 1;
    const int ncol = wv * 16 + l16;
    const float* xr = X + (size_t)row * 128;

    v8f acc = {};
    #pragma unroll
    for (int ko = 0; ko < 128; ko += 32) {
        v16u ahu, alu, bhu, blu;
        const int ka = ko + half * 8;
        #pragma unroll
        for (int j = 0; j < 8; ++j) {
            float f0 = xr[ka + j];
            float f1 = xr[ka + 16 + j];
            if (BN_RELU) {
                f0 = fmaxf(fmaf(f0, scale[ka + j],      shift[ka + j]),      0.0f);
                f1 = fmaxf(fmaf(f1, scale[ka + 16 + j], shift[ka + 16 + j]), 0.0f);
            }
            unsigned short h0 = f2bf(f0);
            ahu[j]     = h0;  alu[j]     = f2bf(f0 - bf2f(h0));
            unsigned short h1 = f2bf(f1);
            ahu[8 + j] = h1;  alu[8 + j] = f2bf(f1 - bf2f(h1));
        }
        const int kb = ko + half * 8;
        #pragma unroll
        for (int j = 0; j < 16; ++j) {
            float w = W[(size_t)(kb + (j & 7) + (j >> 3) * 16) * NCOLS + ncol];
            unsigned short h = f2bf(w);
            bhu[j] = h;  blu[j] = f2bf(w - bf2f(h));
        }
        v16bf ah = __builtin_bit_cast(v16bf, ahu);
        v16bf al = __builtin_bit_cast(v16bf, alu);
        v16bf bh = __builtin_bit_cast(v16bf, bhu);
        v16bf bl = __builtin_bit_cast(v16bf, blu);
        acc = __builtin_amdgcn_wmma_f32_16x16x32_bf16(false, ah, false, bh, (short)0, acc, false, false);
        acc = __builtin_amdgcn_wmma_f32_16x16x32_bf16(false, ah, false, bl, (short)0, acc, false, false);
        acc = __builtin_amdgcn_wmma_f32_16x16x32_bf16(false, al, false, bh, (short)0, acc, false, false);
    }
    const int orow = blockIdx.x * 16 + half * 8;
    #pragma unroll
    for (int i = 0; i < 8; ++i) {
        int r = orow + i;
        if (r < M) H[(size_t)r * NCOLS + wv * 16 + l16] = acc[i];
    }
}

// ---------- launch ----------
extern "C" void kernel_launch(void* const* d_in, const int* in_sizes, int n_in,
                              void* d_out, int out_size, void* d_ws, size_t ws_size,
                              hipStream_t stream) {
    const float* x   = (const float*)d_in[0];
    const float* W1  = (const float*)d_in[1];
    const float* g1  = (const float*)d_in[3];
    const float* be1 = (const float*)d_in[4];
    const float* W2  = (const float*)d_in[5];
    const float* g2  = (const float*)d_in[7];
    const float* be2 = (const float*)d_in[8];
    const int*   ei  = (const int*)d_in[9];

    const int M = in_sizes[0] / 128;   // 100000 nodes
    const int E = in_sizes[9] / 2;     // 1.6M edges
    const int* esrc = ei;
    const int* edst = ei + E;

    // workspace carve-up (256B aligned slices)
    char* p = (char*)d_ws;
    auto take = [&](size_t bytes) {
        char* r = p; p += (bytes + 255) & ~(size_t)255; return r;
    };
    float* dinv  = (float*)take((size_t)M * 4);
    float* B1    = (float*)take((size_t)M * 128 * 4);  // H1, later H2
    float* B2    = (float*)take((size_t)M * 128 * 4);  // A1, later A2
    float* stats = (float*)take(384 * 4);
    float* scsh  = (float*)take(384 * 4);
    float* gsum1 = stats,       *gsq1  = stats + 128;
    float* gsum2 = stats + 256, *gsq2  = stats + 320;
    float* scale1 = scsh,       *shift1 = scsh + 128;
    float* scale2 = scsh + 256, *shift2 = scsh + 320;
    size_t base_need = (size_t)(p - (char*)d_ws);

    // fast-path extras: packed bf16 hi/lo X and swizzled W fragments
    unsigned short* Xhi  = (unsigned short*)take((size_t)M * 128 * 2);
    unsigned short* Xlo  = (unsigned short*)take((size_t)M * 128 * 2);
    unsigned short* Wf1h = (unsigned short*)take(16384 * 2);
    unsigned short* Wf1l = (unsigned short*)take(16384 * 2);
    unsigned short* Wf2h = (unsigned short*)take(8192 * 2);
    unsigned short* Wf2l = (unsigned short*)take(8192 * 2);
    size_t fast_need = (size_t)(p - (char*)d_ws);
    const bool fast = (ws_size >= fast_need) && base_need <= ws_size;

    const int B = 256;
    const int mt = (M + 15) / 16;

    // degrees + dinv
    int imax = (M > 384 ? M : 384);
    k_init  <<<(imax + B - 1) / B, B, 0, stream>>>(dinv, stats, M);
    k_degree<<<(E + B - 1) / B,    B, 0, stream>>>(edst, dinv, E);
    k_dinv  <<<(M + B - 1) / B,    B, 0, stream>>>(dinv, M);

    // layer 1: H1 = X*W1
    if (fast) {
        k_cvtX<false><<<((size_t)M * 128 + B - 1) / B, B, 0, stream>>>(x, nullptr, nullptr, Xhi, Xlo, M * 128);
        k_cvtW<<<(16384 + B - 1) / B, B, 0, stream>>>(W1, Wf1h, Wf1l, 128);
        k_gemm_fast<128><<<mt, 256, 0, stream>>>(Xhi, Xlo, Wf1h, Wf1l, B1, M);
    } else {
        k_gemm<128, false><<<mt, 256, 0, stream>>>(x, W1, B1, nullptr, nullptr, M);
    }
    k_selfinit<128><<<((size_t)M * 32 + B - 1) / B, B, 0, stream>>>(B1, dinv, B2, M);
    k_scatter<128><<<((size_t)E * 32 + B - 1) / B, B, 0, stream>>>(esrc, edst, dinv, B1, B2, E);
    k_stats<128><<<512, 256, 0, stream>>>(B2, gsum1, gsq1, M);
    k_bnfin<<<1, 128, 0, stream>>>(gsum1, gsq1, g1, be1, scale1, shift1, 128, 1.0f / (float)M);

    // layer 2: H2 = relu(bn(A1))*W2
    if (fast) {
        k_cvtX<true><<<((size_t)M * 128 + B - 1) / B, B, 0, stream>>>(B2, scale1, shift1, Xhi, Xlo, M * 128);
        k_cvtW<<<(8192 + B - 1) / B, B, 0, stream>>>(W2, Wf2h, Wf2l, 64);
        k_gemm_fast<64><<<mt, 128, 0, stream>>>(Xhi, Xlo, Wf2h, Wf2l, B1, M);
    } else {
        k_gemm<64, true><<<mt, 128, 0, stream>>>(B2, W2, B1, scale1, shift1, M);
    }
    k_selfinit<64><<<((size_t)M * 16 + B - 1) / B, B, 0, stream>>>(B1, dinv, B2, M);
    k_scatter<64><<<((size_t)E * 16 + B - 1) / B, B, 0, stream>>>(esrc, edst, dinv, B1, B2, E);
    k_stats<64><<<512, 256, 0, stream>>>(B2, gsum2, gsq2, M);
    k_bnfin<<<1, 64, 0, stream>>>(gsum2, gsq2, g2, be2, scale2, shift2, 64, 1.0f / (float)M);

    // final normalize -> d_out (fp32)
    k_out<<<((size_t)M * 64 + B - 1) / B, B, 0, stream>>>(B2, scale2, shift2, (float*)d_out, M);
}
